// VectorQuantize_73392401154080
// MI455X (gfx1250) — compile-verified, matrix operand order checked by reading, not run on hardware
//
#include <hip/hip_runtime.h>
#include <hip/hip_bf16.h>

#define FEATURES 256
#define CODES    8192
#define N_VEC    (16 * 32 * 32)            /* 16384 query vectors        */
#define NZ       (16 * FEATURES * 32 * 32) /* 4194304 elements of z / zq */

typedef __attribute__((ext_vector_type(16))) __bf16 v16bf;
typedef __attribute__((ext_vector_type(8)))  __bf16 v8bf;
typedef __attribute__((ext_vector_type(8)))  float  v8f;

union ABFrag { v16bf v; v8bf h[2]; };

// ---------------- prep: z [B,C,H,W] f32 -> zt [N,C] bf16 (transpose) ----------
__global__ void vq_prep_z(const float* __restrict__ z, __bf16* __restrict__ zt) {
    int i  = blockIdx.x * blockDim.x + threadIdx.x;   // grid == NZ exactly
    int hw = i & 1023;
    int c  = (i >> 10) & 255;
    int b  = i >> 18;
    int n  = (b << 10) | hw;                          // (b,h,w) flattened
    zt[(size_t)n * FEATURES + c] = (__bf16)z[i];
}

// ---------------- prep: codebook f32 -> bf16 ---------------------------------
__global__ void vq_prep_cb(const float* __restrict__ cb, __bf16* __restrict__ cbb) {
    int i = blockIdx.x * blockDim.x + threadIdx.x;    // grid == CODES*FEATURES
    cbb[i] = (__bf16)cb[i];
}

// ---------------- prep: halfnorm[k] = 0.5 * |codebook[k]|^2 ------------------
__global__ void vq_halfnorm(const float* __restrict__ cb, float* __restrict__ hn) {
    int gid  = blockIdx.x * blockDim.x + threadIdx.x;
    int wave = gid >> 5;                              // one wave32 per code
    int lane = threadIdx.x & 31;
    const float* row = cb + (size_t)wave * FEATURES;
    float s = 0.f;
#pragma unroll
    for (int j = 0; j < FEATURES / 32; ++j) { float v = row[lane + 32 * j]; s += v * v; }
#pragma unroll
    for (int m = 16; m >= 1; m >>= 1) s += __shfl_xor(s, m, 32);
    if (lane == 0) hn[wave] = 0.5f * s;
}

// ---------------- main: WMMA GEMM + running argmax ---------------------------
// argmin_k |z-c_k|^2  ==  argmax_k ( z.c_k - 0.5*|c_k|^2 )   (|z|^2 dropped)
#define TC         32    /* codes per iteration (two 16x16 WMMA N-tiles)     */
#define NT         (CODES / TC)            /* 256 iterations                 */
#define LDS_STRIDE 264   /* 256 + 8 bf16 pad: 2-way-per-bank b128 pattern   */

__global__ void __launch_bounds__(128)
vq_argmax(const __bf16* __restrict__ zt, const __bf16* __restrict__ cbb,
          const float* __restrict__ hn, int* __restrict__ idx_out) {
    __shared__ __bf16 buf[2][TC * LDS_STRIDE];        // ping-pong, ~33 KB

    const int tid  = threadIdx.x;
    const int lane = tid & 31;
    const int wave = tid >> 5;        // 4 waves, 16 rows each
    const int half = lane >> 4;       // 16-bit A/B layout: lanes 16..31 hold K+8
    const int l16  = lane & 15;
    const int rowbase = blockIdx.x * 64 + wave * 16;

    // Preload this wave's 16x256 A tile into registers (reused for all 8192 codes).
    ABFrag a[8];
    {
        const v8bf* zr = (const v8bf*)(zt + (size_t)(rowbase + l16) * FEATURES);
#pragma unroll
        for (int c = 0; c < 8; ++c) {        // K chunk c covers K = 32c..32c+31
            a[c].h[0] = zr[4 * c + half];     // K = 32c + 8*half + [0..8)
            a[c].h[1] = zr[4 * c + 2 + half]; // K = 32c + 16 + 8*half + [0..8)
        }
    }

    float bv[8]; int bi[8];
#pragma unroll
    for (int i = 0; i < 8; ++i) { bv[i] = -3.402823466e38f; bi[i] = 0; }

    // Cooperative staging map: 128 threads x 64 bf16 = one 32x256 tile.
    const int ldr = tid >> 2;            // tile row 0..31
    const int ldc = (tid & 3) * 64;      // col start

    // Prologue: stage tile 0 into buffer 0.
    {
        const v8bf* src = (const v8bf*)(cbb + (size_t)ldr * FEATURES + ldc);
        v8bf*       dst = (v8bf*)(&buf[0][ldr * LDS_STRIDE + ldc]);
#pragma unroll
        for (int k = 0; k < 8; ++k) dst[k] = src[k];
    }
    __syncthreads();

    for (int t = 0; t < NT; ++t) {
        const int code0 = t * TC;
        const int p     = t & 1;
        const int tn    = (t + 1) & (NT - 1);   // next tile (wraps harmlessly)

        // Issue next tile's global loads now; WMMAs below hide the latency.
        v8bf stage[8];
        {
            const v8bf* src =
                (const v8bf*)(cbb + (size_t)(tn * TC + ldr) * FEATURES + ldc);
#pragma unroll
            for (int k = 0; k < 8; ++k) stage[k] = src[k];
            const int t2 = (t + 2) & (NT - 1);
            __builtin_prefetch(cbb + (size_t)(t2 * TC + ldr) * FEATURES + ldc, 0, 3);
        }

        // Per-lane code half-norms straight from global (32 KB, cache-resident).
        const float h0 = hn[code0 + l16];
        const float h1 = hn[code0 + 16 + l16];

        // Two independent accumulator chains -> ILP on the tensor pipe.
        v8f acc0 = {0.f, 0.f, 0.f, 0.f, 0.f, 0.f, 0.f, 0.f};
        v8f acc1 = {0.f, 0.f, 0.f, 0.f, 0.f, 0.f, 0.f, 0.f};
        const v8bf* br0 = (const v8bf*)(&buf[p][l16 * LDS_STRIDE]);
        const v8bf* br1 = (const v8bf*)(&buf[p][(16 + l16) * LDS_STRIDE]);
#pragma unroll
        for (int c = 0; c < 8; ++c) {
            ABFrag b0, b1;
            b0.h[0] = br0[4 * c + half];
            b0.h[1] = br0[4 * c + 2 + half];
            b1.h[0] = br1[4 * c + half];
            b1.h[1] = br1[4 * c + 2 + half];
            acc0 = __builtin_amdgcn_wmma_f32_16x16x32_bf16(
                false, a[c].v, false, b0.v, (short)0, acc0, false, false);
            acc1 = __builtin_amdgcn_wmma_f32_16x16x32_bf16(
                false, a[c].v, false, b1.v, (short)0, acc1, false, false);
        }

        // Drain staged tile into the other buffer (waits on loadcnt here, late).
        {
            v8bf* dst = (v8bf*)(&buf[1 - p][ldr * LDS_STRIDE + ldc]);
#pragma unroll
            for (int k = 0; k < 8; ++k) dst[k] = stage[k];
        }

        // Running argmax update; acc[i]: row = rowbase + i + 8*half.
        const int c0 = code0 + l16;
        const int c1 = code0 + 16 + l16;
#pragma unroll
        for (int i = 0; i < 8; ++i) {
            float s0 = acc0[i] - h0;
            if (s0 > bv[i]) { bv[i] = s0; bi[i] = c0; }
            float s1 = acc1[i] - h1;
            if (s1 > bv[i]) { bv[i] = s1; bi[i] = c1; }
        }

        __syncthreads();   // single barrier: publishes buf[1-p], guards reuse
    }

    // Argmax across the 16 lanes of each half (codes were lane-partitioned).
#pragma unroll
    for (int i = 0; i < 8; ++i) {
        float v = bv[i]; int ix = bi[i];
#pragma unroll
        for (int m = 8; m >= 1; m >>= 1) {   // stays inside each 16-lane half
            float ov = __shfl_xor(v, m, 32);
            int   oi = __shfl_xor(ix, m, 32);
            if (ov > v || (ov == v && oi < ix)) { v = ov; ix = oi; }
        }
        if (l16 == 0) idx_out[rowbase + i + 8 * half] = ix;
    }
}

// ---------------- gather: zq[b,c,h,w] = codebook_f32[idx[n], c] --------------
__global__ void vq_gather(const float* __restrict__ cb, const int* __restrict__ idx,
                          float* __restrict__ zq) {
    int o  = blockIdx.x * blockDim.x + threadIdx.x;   // grid == NZ exactly
    int hw = o & 1023;
    int c  = (o >> 10) & 255;
    int b  = o >> 18;
    int n  = (b << 10) | hw;
    zq[o] = cb[(size_t)idx[n] * FEATURES + c];
}

extern "C" void kernel_launch(void* const* d_in, const int* in_sizes, int n_in,
                              void* d_out, int out_size, void* d_ws, size_t ws_size,
                              hipStream_t stream) {
    const float* z  = (const float*)d_in[0];   // [16,256,32,32]
    const float* cb = (const float*)d_in[1];   // [8192,256]

    float* zq  = (float*)d_out;                // output 0: zq [16,256,32,32] f32
    int*   idx = (int*)d_out + NZ;             // output 1: idx [16,32,32] int32

    // workspace layout: zt bf16 (8MB) | cb bf16 (4MB) | halfnorm f32 (32KB)
    __bf16* zt  = (__bf16*)d_ws;
    __bf16* cbb = (__bf16*)((char*)d_ws + (size_t)(8u << 20));
    float*  hn  = (float*)((char*)d_ws + (size_t)(12u << 20));

    vq_prep_z  <<<NZ / 256, 256, 0, stream>>>(z, zt);
    vq_prep_cb <<<(CODES * FEATURES) / 256, 256, 0, stream>>>(cb, cbb);
    vq_halfnorm<<<CODES / 8, 256, 0, stream>>>(cb, hn);
    vq_argmax  <<<N_VEC / 64, 128, 0, stream>>>(zt, cbb, hn, idx);
    vq_gather  <<<NZ / 256, 256, 0, stream>>>(cb, idx, zq);
}
